// LSTMLayer_90640989815369
// MI455X (gfx1250) — matrix operand order repaired and checked by reading, not compiled, not run on hardware
//
#include <hip/hip_runtime.h>
#include <hip/hip_bf16.h>
#include <stdint.h>

typedef __attribute__((ext_vector_type(16))) __bf16 v16bf;
typedef __attribute__((ext_vector_type(8)))  __bf16 v8bf;
typedef __attribute__((ext_vector_type(8)))  float  v8f;
typedef __attribute__((ext_vector_type(4)))  unsigned int v4u;
typedef __attribute__((ext_vector_type(8)))  int    v8i;
typedef __attribute__((ext_vector_type(4)))  int    v4i;

#define T_STEPS 2048
#define BATCH   32
#define DIM     512           // D == N == 512
#define KT_N    16            // 512 / 32 k-steps
#define NT_N    32            // 512 / 16 column tiles

// ---------------- helpers ----------------
__device__ __forceinline__ v8f vzero8() {
    v8f z;
#pragma unroll
    for (int i = 0; i < 8; ++i) z[i] = 0.0f;
    return z;
}

__device__ __forceinline__ v16bf mk16(v8bf lo, v8bf hi) {
    v16bf r;
#pragma unroll
    for (int i = 0; i < 8; ++i) { r[i] = lo[i]; r[i + 8] = hi[i]; }
    return r;
}

__device__ __forceinline__ float sigmoidf_(float x) {
    return 1.0f / (1.0f + __expf(-x));
}

// grid-wide sense-reversal barrier (32 co-resident persistent WGs)
__device__ __forceinline__ void grid_barrier(unsigned* bar, unsigned nblocks) {
    __syncthreads();
    if (threadIdx.x == 0) {
        __threadfence();
        unsigned gen = __hip_atomic_load(bar + 1, __ATOMIC_RELAXED, __HIP_MEMORY_SCOPE_AGENT);
        unsigned arr = __hip_atomic_fetch_add(bar, 1u, __ATOMIC_ACQ_REL, __HIP_MEMORY_SCOPE_AGENT);
        if (arr == nblocks - 1u) {
            __hip_atomic_store(bar, 0u, __ATOMIC_RELAXED, __HIP_MEMORY_SCOPE_AGENT);
            __hip_atomic_store(bar + 1, gen + 1u, __ATOMIC_RELEASE, __HIP_MEMORY_SCOPE_AGENT);
        } else {
            while (__hip_atomic_load(bar + 1, __ATOMIC_ACQUIRE, __HIP_MEMORY_SCOPE_AGENT) == gen)
                __builtin_amdgcn_s_sleep(2);
        }
    }
    __syncthreads();
}

// ---------------- x -> bf16 one-time conversion ----------------
// x[B][T][D] f32 -> xbf[B][T][D] bf16, 8 elements per thread.
__global__ void __launch_bounds__(256)
convert_x(const float* __restrict__ x, __bf16* __restrict__ xbf) {
    size_t i = ((size_t)blockIdx.x * 256 + threadIdx.x) * 8;
    v8f v = *(const v8f*)(x + i);
    v8bf o;
#pragma unroll
    for (int j = 0; j < 8; ++j) o[j] = (__bf16)v[j];
    *(v8bf*)(xbf + i) = o;
}

// ---------------- weight packing ----------------
// Pack a 512x512 row-major f32 matrix into bf16 WMMA B-fragments.
// Fragment (nt, kt): lane l holds 16 values: n = nt*16 + (l&15),
// k = kt*32 + (l>>4)*8 + (e<8 ? e : e+8).  dst frag index = nt*ntStrideFrags + kt.
__global__ void pack_b_frags(const float* __restrict__ src,
                             __bf16* __restrict__ dst, int ntStrideFrags) {
    int tid  = blockIdx.x * 256 + threadIdx.x;   // NT*KT*32 = 16384 threads
    int lane = tid & 31;
    int kt   = (tid >> 5) & 15;
    int nt   = tid >> 9;
    int n    = nt * 16 + (lane & 15);
    int kb   = kt * 32 + (lane >> 4) * 8;
    __bf16* out = dst + ((size_t)nt * ntStrideFrags + kt) * 512 + lane * 16;
#pragma unroll
    for (int e = 0; e < 16; ++e)
        out[e] = (__bf16)src[(size_t)(kb + (e < 8 ? e : e + 8)) * DIM + n];
}

// ---------------- phase 1: x @ [W_i|W_f|W_g|W_c] + b ----------------
// A rows r = t*32 + b of xbf[B][T][D]; output xacts[4][T][B][N] in bf16.
__global__ void __launch_bounds__(256)
xproj_gemm(const __bf16* __restrict__ xbf,
           const __bf16* __restrict__ packedW,   // [4][32 nt][16 kt][32 lane][16]
           const float* __restrict__ b_i, const float* __restrict__ b_f,
           const float* __restrict__ b_g, const float* __restrict__ b_c,
           __bf16* __restrict__ xacts) {
    const int lane   = threadIdx.x & 31;
    const int wid    = blockIdx.x * 8 + (threadIdx.x >> 5);
    const int mblock = wid >> 7;          // 64-row blocks (1024 total)
    const int ncomb  = wid & 127;         // gate*32 + nt
    const int g      = ncomb >> 5;
    const int nt     = ncomb & 31;
    const int hi8    = (lane >> 4) * 8;
    const int mrow   = lane & 15;

    const __bf16* bfr = packedW + (((size_t)g * NT_N + nt) * KT_N) * 512 + lane * 16;

    // per-mi A row base pointers (row r = t*32 + b maps to xbf[b][t][:])
    const __bf16* arow[4];
#pragma unroll
    for (int mi = 0; mi < 4; ++mi) {
        int r = mblock * 64 + mi * 16 + mrow;
        int t = r >> 5, b = r & 31;
        arow[mi] = xbf + ((size_t)b * T_STEPS + t) * DIM + hi8;
    }

    v8f acc[4];
#pragma unroll
    for (int mi = 0; mi < 4; ++mi) acc[mi] = vzero8();

#pragma unroll 4
    for (int kt = 0; kt < KT_N; ++kt) {
        v16bf bf = *(const v16bf*)(bfr + (size_t)kt * 512);
#pragma unroll
        for (int mi = 0; mi < 4; ++mi) {
            const __bf16* ap = arow[mi] + kt * 32;
            v8bf lo = *(const v8bf*)ap;
            v8bf hh = *(const v8bf*)(ap + 16);
            acc[mi] = __builtin_amdgcn_wmma_f32_16x16x32_bf16(
                false, mk16(lo, hh), false, bf, (short)0, acc[mi], false, false);
        }
    }

    const float* bp = (g == 0) ? b_i : (g == 1) ? b_f : (g == 2) ? b_g : b_c;
    const int n = nt * 16 + (lane & 15);
    const float bias = bp[n];
#pragma unroll
    for (int mi = 0; mi < 4; ++mi)
#pragma unroll
        for (int j = 0; j < 8; ++j) {
            int r = mblock * 64 + mi * 16 + j + hi8;
            int t = r >> 5, b = r & 31;
            xacts[(((size_t)g * T_STEPS + t) * BATCH + b) * DIM + n] =
                (__bf16)(acc[mi][j] + bias);
        }
}

// ---------------- phase 2: persistent recurrence ----------------
// 32 WGs x 256 threads. WG owns N-columns [wg*16, wg*16+16) for all 4 gates.
// U slice (64KB, fragment-packed) -> LDS via Tensor Data Mover, once.
__global__ void __launch_bounds__(256)
lstm_recurrence(const __bf16* __restrict__ packedU,   // [32 wg][4 g][16 kt][32][16]
                const __bf16* __restrict__ xacts,     // [4][T][B][N] bf16
                __bf16* __restrict__ hbuf,            // [2][32][512] bf16 (zeroed)
                __bf16* __restrict__ hs,              // [T][B][N] bf16
                unsigned* __restrict__ bar) {
    extern __shared__ char smem[];                    // 65536 B: packed U frags
    __shared__ float gateBuf[4 * BATCH * 16];         // gate activations, f32
    __bf16* Ulds = (__bf16*)smem;

    const int tid   = threadIdx.x;
    const int lane  = tid & 31;
    const int w     = tid >> 5;
    const int g     = w >> 1;        // gate 0..3
    const int mtile = w & 1;         // batch half 0..1
    const int wgN   = blockIdx.x * 16;
    const int hi8   = (lane >> 4) * 8;

    // ---- TDM: DMA this WG's 64KB packed-U slice into LDS (1D tile, 8B elems)
    if (w == 0) {
        uint64_t gaddr  = (uint64_t)(uintptr_t)packedU + (uint64_t)blockIdx.x * 65536u;
        uint32_t ldsoff = (uint32_t)(uintptr_t)(void*)Ulds;   // low 32b == LDS offset
        v4u g0; v8i g1; v4i g2; v4i g3; v8i g4;
        g0[0] = 1u;                                    // count=1, user desc
        g0[1] = ldsoff;                                // lds_addr
        g0[2] = (uint32_t)gaddr;                       // global_addr[31:0]
        g0[3] = (uint32_t)((gaddr >> 32) & 0x01FFFFFFu) | (2u << 30);  // type=2
        g1[0] = (int)(3u << 16);                       // data_size=8B; mask=0
        g1[1] = (int)((8192u & 0xFFFFu) << 16);        // tensor_dim0 = 8192 (lo16)
        g1[2] = (int)((8192u >> 16) | (1u << 16));     // dim0 hi16 | tensor_dim1=1
        g1[3] = (int)(8192u << 16);                    // tile_dim0 = 8192 elems
        g1[4] = 0;                                     // tile_dim1/2 unused
        g1[5] = (int)8192u;                            // tensor_dim0_stride lo32
        g1[6] = 0; g1[7] = 0;
        g2[0] = g2[1] = g2[2] = g2[3] = 0;
        g3[0] = g3[1] = g3[2] = g3[3] = 0;
        g4[0] = g4[1] = g4[2] = g4[3] = 0;
        g4[4] = g4[5] = g4[6] = g4[7] = 0;
        // clang-23 / therock-10.0 6-arg form: (g0, g1, g2, g3, g4, cpol)
        __builtin_amdgcn_tensor_load_to_lds(g0, g1, g2, g3, g4, 0);
        __builtin_amdgcn_s_wait_tensorcnt(0);
    }
    __syncthreads();

    float cst0 = 0.0f, cst1 = 0.0f;   // persistent cell state: 2 cells/thread
    const int cell0 = tid, cell1 = tid + 256;

    for (int t = 0; t < T_STEPS; ++t) {
        const __bf16* cur = hbuf + (size_t)(t & 1) * (BATCH * DIM);
        __bf16*       nxt = hbuf + (size_t)((t & 1) ^ 1) * (BATCH * DIM);

        // acc = h(t) @ U_g  for this wave's (gate, batch-half) 16x16 tile
        v8f acc = vzero8();
#pragma unroll 4
        for (int kt = 0; kt < KT_N; ++kt) {
            const __bf16* ap = cur + (size_t)(mtile * 16 + (lane & 15)) * DIM + kt * 32 + hi8;
            v8bf lo = *(const v8bf*)ap;
            v8bf hh = *(const v8bf*)(ap + 16);
            v16bf uf = *(const v16bf*)(Ulds + ((size_t)(g * KT_N + kt) * 32 + lane) * 16);
            acc = __builtin_amdgcn_wmma_f32_16x16x32_bf16(
                false, mk16(lo, hh), false, uf, (short)0, acc, false, false);
        }

        // + x preactivation, apply gate nonlinearity, deposit into LDS
        {
            const int n = wgN + (lane & 15);
#pragma unroll
            for (int j = 0; j < 8; ++j) {
                int b = mtile * 16 + j + hi8;
                float v = acc[j] +
                    (float)xacts[(((size_t)g * T_STEPS + t) * BATCH + b) * DIM + n];
                float a = (g == 3) ? tanhf(v) : sigmoidf_(v);   // i,f,g: sigmoid; c~: tanh
                gateBuf[(g * BATCH + b) * 16 + (lane & 15)] = a;
            }
        }
        __syncthreads();

        // cell update: each thread owns 2 (b, n) cells
        {
            int b0 = cell0 >> 4, n0 = cell0 & 15;
            float iv = gateBuf[(0 * BATCH + b0) * 16 + n0];
            float fv = gateBuf[(1 * BATCH + b0) * 16 + n0];
            float gv = gateBuf[(2 * BATCH + b0) * 16 + n0];
            float cv = gateBuf[(3 * BATCH + b0) * 16 + n0];
            cst0 = fv * cst0 + iv * cv;
            __bf16 h0 = (__bf16)(gv * tanhf(cst0));
            nxt[(size_t)b0 * DIM + wgN + n0] = h0;
            hs[((size_t)t * BATCH + b0) * DIM + wgN + n0] = h0;

            int b1 = cell1 >> 4, n1 = cell1 & 15;
            iv = gateBuf[(0 * BATCH + b1) * 16 + n1];
            fv = gateBuf[(1 * BATCH + b1) * 16 + n1];
            gv = gateBuf[(2 * BATCH + b1) * 16 + n1];
            cv = gateBuf[(3 * BATCH + b1) * 16 + n1];
            cst1 = fv * cst1 + iv * cv;
            __bf16 h1 = (__bf16)(gv * tanhf(cst1));
            nxt[(size_t)b1 * DIM + wgN + n1] = h1;
            hs[((size_t)t * BATCH + b1) * DIM + wgN + n1] = h1;
        }
        grid_barrier(bar, gridDim.x);
    }
}

// ---------------- phase 3: relu(hs @ W_o + b_o) -> out[B][T][N] f32 ----------------
__global__ void __launch_bounds__(256)
out_gemm(const __bf16* __restrict__ hs,        // [T][B][N] bf16
         const __bf16* __restrict__ packedWo,  // [32 nt][16 kt][32][16]
         const float* __restrict__ b_o,
         float* __restrict__ out) {
    const int lane   = threadIdx.x & 31;
    const int wid    = blockIdx.x * 8 + (threadIdx.x >> 5);
    const int mblock = wid >> 5;          // 1024 blocks of 64 rows
    const int nt     = wid & 31;
    const int hi8    = (lane >> 4) * 8;
    const int mrow   = lane & 15;

    const __bf16* bfr = packedWo + ((size_t)nt * KT_N) * 512 + lane * 16;

    v8f acc[4];
#pragma unroll
    for (int mi = 0; mi < 4; ++mi) acc[mi] = vzero8();

#pragma unroll 4
    for (int kt = 0; kt < KT_N; ++kt) {
        v16bf bf = *(const v16bf*)(bfr + (size_t)kt * 512);
#pragma unroll
        for (int mi = 0; mi < 4; ++mi) {
            size_t r = (size_t)mblock * 64 + mi * 16 + mrow;
            const __bf16* ap = hs + r * DIM + kt * 32 + hi8;
            v8bf lo = *(const v8bf*)ap;
            v8bf hh = *(const v8bf*)(ap + 16);
            acc[mi] = __builtin_amdgcn_wmma_f32_16x16x32_bf16(
                false, mk16(lo, hh), false, bf, (short)0, acc[mi], false, false);
        }
    }

    const int n = nt * 16 + (lane & 15);
    const float bias = b_o[n];
#pragma unroll
    for (int mi = 0; mi < 4; ++mi)
#pragma unroll
        for (int j = 0; j < 8; ++j) {
            int r = mblock * 64 + mi * 16 + j + hi8;
            int t = r >> 5, b = r & 31;
            float v = acc[mi][j] + bias;
            out[((size_t)b * T_STEPS + t) * DIM + n] = v > 0.0f ? v : 0.0f;
        }
}

// ---------------- launch ----------------
extern "C" void kernel_launch(void* const* d_in, const int* in_sizes, int n_in,
                              void* d_out, int out_size, void* d_ws, size_t ws_size,
                              hipStream_t stream) {
    const float* x   = (const float*)d_in[0];
    const float* W_i = (const float*)d_in[1];
    const float* U_i = (const float*)d_in[2];
    const float* W_f = (const float*)d_in[3];
    const float* U_f = (const float*)d_in[4];
    const float* W_g = (const float*)d_in[5];
    const float* U_g = (const float*)d_in[6];
    const float* W_c = (const float*)d_in[7];
    const float* U_c = (const float*)d_in[8];
    const float* W_o = (const float*)d_in[9];
    const float* b_i = (const float*)d_in[10];
    const float* b_f = (const float*)d_in[11];
    const float* b_g = (const float*)d_in[12];
    const float* b_c = (const float*)d_in[13];
    const float* b_o = (const float*)d_in[14];

    char* ws = (char*)d_ws;
    size_t off = 0;
    unsigned* bar      = (unsigned*)(ws + off); off += 256;
    __bf16*  hbuf      = (__bf16*)(ws + off);   off += (size_t)2 * BATCH * DIM * 2;      // 64 KB
    __bf16*  packedWin = (__bf16*)(ws + off);   off += (size_t)4 * DIM * DIM * 2;        // 2 MB
    __bf16*  packedU   = (__bf16*)(ws + off);   off += (size_t)4 * DIM * DIM * 2;        // 2 MB
    __bf16*  packedWo  = (__bf16*)(ws + off);   off += (size_t)DIM * DIM * 2;            // 0.5 MB
    __bf16*  xbf       = (__bf16*)(ws + off);   off += (size_t)BATCH * T_STEPS * DIM * 2;     // 67 MB
    __bf16*  xacts     = (__bf16*)(ws + off);   off += (size_t)4 * T_STEPS * BATCH * DIM * 2; // 268 MB
    __bf16*  hs        = (__bf16*)(ws + off);   off += (size_t)T_STEPS * BATCH * DIM * 2;     // 67 MB
    (void)ws_size; (void)in_sizes; (void)n_in; (void)out_size;

    // zero barrier + h double-buffer (graph-capture-legal memset node)
    (void)hipMemsetAsync(d_ws, 0, 256 + (size_t)2 * BATCH * DIM * 2, stream);

    // one-time x -> bf16 (streams 134MB in + 67MB out, ~10us at 23.3TB/s)
    convert_x<<<16384, 256, 0, stream>>>(x, xbf);   // 32*2048*512 / 8 / 256

    // pack input-projection weights: layout [g][nt][kt], ntStride = 16 frags
    const size_t WgSz = (size_t)NT_N * KT_N * 512;  // frags-elements per gate
    pack_b_frags<<<64, 256, 0, stream>>>(W_i, packedWin + 0 * WgSz, KT_N);
    pack_b_frags<<<64, 256, 0, stream>>>(W_f, packedWin + 1 * WgSz, KT_N);
    pack_b_frags<<<64, 256, 0, stream>>>(W_g, packedWin + 2 * WgSz, KT_N);
    pack_b_frags<<<64, 256, 0, stream>>>(W_c, packedWin + 3 * WgSz, KT_N);
    // pack recurrent weights: layout [nt][g][kt], so each WG slice is one
    // contiguous 64KB block for the TDM copy; ntStride = 4*16 frags
    pack_b_frags<<<64, 256, 0, stream>>>(U_i, packedU + 0 * (size_t)KT_N * 512, 64);
    pack_b_frags<<<64, 256, 0, stream>>>(U_f, packedU + 1 * (size_t)KT_N * 512, 64);
    pack_b_frags<<<64, 256, 0, stream>>>(U_g, packedU + 2 * (size_t)KT_N * 512, 64);
    pack_b_frags<<<64, 256, 0, stream>>>(U_c, packedU + 3 * (size_t)KT_N * 512, 64);
    pack_b_frags<<<64, 256, 0, stream>>>(W_o, packedWo, KT_N);

    // phase 1: 1024 m-blocks x 128 (gate,ntile) = 131072 waves / 8 per WG
    xproj_gemm<<<16384, 256, 0, stream>>>(xbf, packedWin, b_i, b_f, b_g, b_c, xacts);

    // phase 2: persistent 32-WG recurrence, 64KB dynamic LDS for U fragments
    lstm_recurrence<<<32, 256, 65536, stream>>>(packedU, xacts, hbuf, hs, bar);

    // phase 3: 1024 m-blocks x 32 ntiles = 32768 waves / 8 per WG
    out_gemm<<<4096, 256, 0, stream>>>(hs, packedWo, b_o, (float*)d_out);
}
